// Attention_78580721648014
// MI455X (gfx1250) — matrix operand order repaired and checked by reading, not compiled, not run on hardware
//
#include <hip/hip_runtime.h>

// ---- problem constants --------------------------------------------------
#define DIMX    1024
#define HEADS   16
#define DIM_HEAD 64
#define INNERX  1024           // HEADS*DIM_HEAD
#define BATCH   2
#define SEQ     2048
#define ROWS    (BATCH*SEQ)    // 4096
#define QKV_N   (3*INNERX)     // 3072

typedef __attribute__((ext_vector_type(16))) _Float16 v16h;
typedef __attribute__((ext_vector_type(8)))  _Float16 v8h;
typedef __attribute__((ext_vector_type(8)))  float    v8f;
typedef __attribute__((ext_vector_type(4)))  unsigned int u32x4;
typedef __attribute__((ext_vector_type(8)))  unsigned int u32x8;

union AF { v16h v; v8h h[2]; };

static __device__ __forceinline__ v8f wmma16x16x32(v16h a, v16h b, v8f c) {
  // (neg_a, A, neg_b, B, c_mod, C, reuse_a, reuse_b)
  return __builtin_amdgcn_wmma_f32_16x16x32_f16(false, a, false, b, (short)0, c,
                                                false, false);
}

// ---- TDM: 2D f16 tile load, global -> LDS, with per-row LDS padding -----
// Pads 8 DWORDs (32B = 16 halves) after every 32 DWORDs (128B = 64 halves)
// -> LDS row stride of 80 halves, matching the attention K-tile layout.
static __device__ __forceinline__ void tdm_load_k_tile(const _Float16* gptr,
                                                       unsigned int ldsAddr,
                                                       unsigned int tensorD1) {
  const unsigned long long ga = (unsigned long long)(size_t)gptr;
  u32x4 g0;
  g0[0] = 1u;                                       // count=1, no gather
  g0[1] = ldsAddr;                                  // lds_addr (bytes)
  g0[2] = (unsigned int)(ga & 0xFFFFFFFFull);       // global_addr[31:0]
  g0[3] = (unsigned int)((ga >> 32) & 0x1FFFFFFull) // global_addr[56:32]
          | (2u << 30);                             // type=2 ("image")
  u32x8 g1;
  // [15:0] wg_mask=0 | [17:16] data_size=1 (2B) | [20] pad_enable
  // [24:22] pad_interval=4 (32 DW) | [31:25] pad_amount=7 (8 DW)
  g1[0] = (1u << 16) | (1u << 20) | (4u << 22) | (7u << 25);
  const unsigned int tensorD0 = DIM_HEAD;           // 64 elements per row
  g1[1] = (tensorD0 & 0xFFFFu) << 16;               // dim0[15:0]; atomic_addr=0
  g1[2] = ((tensorD0 >> 16) & 0xFFFFu) | ((tensorD1 & 0xFFFFu) << 16);
  g1[3] = ((tensorD1 >> 16) & 0xFFFFu) | ((unsigned int)DIM_HEAD << 16); // tile_dim0=64
  g1[4] = 64u;                                      // tile_dim1=64, tile_dim2=0
  g1[5] = (unsigned int)DIM_HEAD;                   // dim0_stride[31:0] = 64 elem
  g1[6] = 0u;                                       // dim0_stride[47:32], dim1_stride lo
  g1[7] = 0u;                                       // dim1_stride (unused for 2D)
  asm volatile("tensor_load_to_lds %0, %1" :: "s"(g0), "s"(g1) : "memory");
}

// ---- kernel 1: f32 -> f16 conversion ------------------------------------
__global__ void f32_to_f16_kernel(const float* __restrict__ src,
                                  _Float16* __restrict__ dst, int n) {
  int i = blockIdx.x * blockDim.x + threadIdx.x;
  int stride = gridDim.x * blockDim.x;
  for (; i < n; i += stride) dst[i] = (_Float16)src[i];
}

// ---- kernel 2: LayerNorm -> f16 (one row per block, 256 threads) --------
__global__ __launch_bounds__(256)
void ln_kernel(const float* __restrict__ x, const float* __restrict__ g,
               const float* __restrict__ bb, _Float16* __restrict__ xn) {
  __shared__ float s0[256];
  __shared__ float s1[256];
  const int row = blockIdx.x;
  const int t = threadIdx.x;
  const float4 xv = reinterpret_cast<const float4*>(x + (size_t)row * DIMX)[t];
  float sum = xv.x + xv.y + xv.z + xv.w;
  float sq  = xv.x * xv.x + xv.y * xv.y + xv.z * xv.z + xv.w * xv.w;
  s0[t] = sum; s1[t] = sq;
  __syncthreads();
  for (int off = 128; off > 0; off >>= 1) {
    if (t < off) { s0[t] += s0[t + off]; s1[t] += s1[t + off]; }
    __syncthreads();
  }
  const float mean = s0[0] * (1.0f / DIMX);
  const float var  = s1[0] * (1.0f / DIMX) - mean * mean;
  const float rstd = rsqrtf(var + 1e-5f);
  const int c = t * 4;
  _Float16* out = xn + (size_t)row * DIMX + c;
  out[0] = (_Float16)((xv.x - mean) * rstd * g[c + 0] + bb[c + 0]);
  out[1] = (_Float16)((xv.y - mean) * rstd * g[c + 1] + bb[c + 1]);
  out[2] = (_Float16)((xv.z - mean) * rstd * g[c + 2] + bb[c + 2]);
  out[3] = (_Float16)((xv.w - mean) * rstd * g[c + 3] + bb[c + 3]);
}

// ---- kernel 3: QKV GEMM  [4096,1024]x[1024,3072] -> q/k/v [B,H,N,Dh] ----
__global__ __launch_bounds__(256)
void qkv_gemm_kernel(const _Float16* __restrict__ A,
                     const _Float16* __restrict__ Wh,
                     const float* __restrict__ bias,
                     _Float16* __restrict__ Q, _Float16* __restrict__ K,
                     _Float16* __restrict__ V) {
  __shared__ _Float16 ldsB[64 * 40];            // [n][k], stride 40 halves (80B)
  const int t = threadIdx.x;
  const int lane = t & 31, wave = t >> 5;
  const int halfSel = (lane & 16) ? 1 : 0;
  const int c16 = lane & 15;
  const int colBase = blockIdx.x * 64;
  const int rowBase = blockIdx.y * 128;

  v8f acc[4];
  for (int nt = 0; nt < 4; ++nt)
    for (int i = 0; i < 8; ++i) acc[nt][i] = 0.0f;

  const _Float16* aRow = A + (size_t)(rowBase + wave * 16 + c16) * DIMX;

  for (int kk = 0; kk < DIMX; kk += 32) {
    __syncthreads();
    {   // stage 32x64 B tile transposed: ldsB[n][k]
      const int kr = t & 31;
      const int n8 = (t >> 5) * 8;
      const v8h gv = *reinterpret_cast<const v8h*>(
          Wh + (size_t)(kk + kr) * QKV_N + colBase + n8);
#pragma unroll
      for (int i = 0; i < 8; ++i) ldsB[(n8 + i) * 40 + kr] = gv[i];
    }
    __syncthreads();
    if (kk + 32 < DIMX)
      __builtin_prefetch(Wh + (size_t)(kk + 32 + (t & 31)) * QKV_N + colBase +
                             (t >> 5) * 8, 0, 1);

    AF a;
    a.h[0] = *reinterpret_cast<const v8h*>(aRow + kk + halfSel * 8);
    a.h[1] = *reinterpret_cast<const v8h*>(aRow + kk + 16 + halfSel * 8);
#pragma unroll
    for (int nt = 0; nt < 4; ++nt) {
      AF b;
      const _Float16* bp = ldsB + (nt * 16 + c16) * 40 + halfSel * 8;
      b.h[0] = *reinterpret_cast<const v8h*>(bp);
      b.h[1] = *reinterpret_cast<const v8h*>(bp + 16);
      acc[nt] = wmma16x16x32(a.v, b.v, acc[nt]);
    }
  }

  // epilogue: bias + scatter to [B,H,N,Dh]
  const int which = colBase >> 10;             // 0=q 1=k 2=v (tile never crosses)
  const int head = (colBase & 1023) >> 6;      // constant within tile
  _Float16* dst = (which == 0) ? Q : (which == 1) ? K : V;
#pragma unroll
  for (int nt = 0; nt < 4; ++nt) {
    const float bv = bias[colBase + nt * 16 + c16];
    const int dh = nt * 16 + c16;
#pragma unroll
    for (int r = 0; r < 8; ++r) {
      const int row = rowBase + wave * 16 + r + halfSel * 8;
      const int bidx = row >> 11, n = row & 2047;
      dst[((size_t)(bidx * HEADS + head) * SEQ + n) * DIM_HEAD + dh] =
          (_Float16)(acc[nt][r] + bv);
    }
  }
}

// ---- kernel 4: flash attention, one (b,h), 128 q-rows per block ---------
__global__ __launch_bounds__(256)
void attn_kernel(const _Float16* __restrict__ Q, const _Float16* __restrict__ K,
                 const _Float16* __restrict__ V, _Float16* __restrict__ O) {
  __shared__ _Float16 ldsK[64 * 80];      // [key][d] stride 80 (TDM-padded)
  __shared__ _Float16 ldsV[64 * 80];      // [d][key]   (transposed)
  __shared__ _Float16 ldsP[8 * 16 * 80];  // per-wave P scratch [m][key]
  const int t = threadIdx.x, lane = t & 31, wave = t >> 5;
  const int halfSel = (lane & 16) ? 1 : 0;
  const int c16 = lane & 15;
  const int bh = blockIdx.y;                       // b*16 + h
  const size_t headOff = (size_t)bh * SEQ * DIM_HEAD;
  const int qBase = blockIdx.x * 128 + wave * 16;
  const unsigned int ldsKAddr = (unsigned int)(size_t)(void*)ldsK;

  // Q A-fragments stay in registers for the whole kernel
  const _Float16* qRow = Q + headOff + (size_t)(qBase + c16) * DIM_HEAD;
  AF aq[2];
  aq[0].h[0] = *reinterpret_cast<const v8h*>(qRow + halfSel * 8);
  aq[0].h[1] = *reinterpret_cast<const v8h*>(qRow + 16 + halfSel * 8);
  aq[1].h[0] = *reinterpret_cast<const v8h*>(qRow + 32 + halfSel * 8);
  aq[1].h[1] = *reinterpret_cast<const v8h*>(qRow + 48 + halfSel * 8);

  float mrun[8], lrun[8];
  v8f oacc[4];
#pragma unroll
  for (int r = 0; r < 8; ++r) { mrun[r] = -3.0e38f; lrun[r] = 0.0f; }
  for (int nt = 0; nt < 4; ++nt)
    for (int i = 0; i < 8; ++i) oacc[nt][i] = 0.0f;

  _Float16* pW   = ldsP + wave * 16 * 80;               // write base
  const _Float16* pRow = ldsP + (wave * 16 + c16) * 80; // A-frag read row

  for (int kt = 0; kt < SEQ; kt += 64) {
    __syncthreads();
    // K tile: one Tensor Data Mover op (wave-uniform; EXEC ignored by TDM),
    // hardware pads each 64-half row by 16 halves -> LDS stride 80.
    if (wave == 0) {
      tdm_load_k_tile(K + headOff + (size_t)kt * DIM_HEAD, ldsKAddr,
                      (unsigned int)(SEQ - kt));
      __builtin_amdgcn_s_wait_tensorcnt(0);
    }
    {   // stage 64x64 V transposed: ldsV[d][key]
      const int key = t >> 2;
      const int d8  = (t & 3) * 16;
      const _Float16* vg = V + headOff + (size_t)(kt + key) * DIM_HEAD + d8;
      const v8h v0 = *reinterpret_cast<const v8h*>(vg);
      const v8h v1 = *reinterpret_cast<const v8h*>(vg + 8);
#pragma unroll
      for (int i = 0; i < 8; ++i) ldsV[(d8 + i) * 80 + key] = v0[i];
#pragma unroll
      for (int i = 0; i < 8; ++i) ldsV[(d8 + 8 + i) * 80 + key] = v1[i];
    }
    __syncthreads();

    // S = Q * K^T  (16 rows x 64 keys per wave)
    v8f s[4];
    for (int nt = 0; nt < 4; ++nt)
      for (int i = 0; i < 8; ++i) s[nt][i] = 0.0f;
#pragma unroll
    for (int kc = 0; kc < 2; ++kc) {
#pragma unroll
      for (int nt = 0; nt < 4; ++nt) {
        AF b;
        const _Float16* bp = ldsK + (nt * 16 + c16) * 80 + kc * 32 + halfSel * 8;
        b.h[0] = *reinterpret_cast<const v8h*>(bp);
        b.h[1] = *reinterpret_cast<const v8h*>(bp + 16);
        s[nt] = wmma16x16x32(aq[kc].v, b.v, s[nt]);
      }
    }

    // online softmax (row stats via shfl_xor across 16-lane C-frag halves)
    const float scale = 0.125f;                  // 1/sqrt(64)
#pragma unroll
    for (int r = 0; r < 8; ++r) {
      float mx = -3.0e38f;
#pragma unroll
      for (int nt = 0; nt < 4; ++nt) {
        const float sv = s[nt][r] * scale;
        s[nt][r] = sv;
        mx = fmaxf(mx, sv);
      }
      mx = fmaxf(mx, __shfl_xor(mx, 1, 32));
      mx = fmaxf(mx, __shfl_xor(mx, 2, 32));
      mx = fmaxf(mx, __shfl_xor(mx, 4, 32));
      mx = fmaxf(mx, __shfl_xor(mx, 8, 32));
      const float newm = fmaxf(mrun[r], mx);
      const float corr = __expf(mrun[r] - newm);
      mrun[r] = newm;
      float rs = 0.0f;
#pragma unroll
      for (int nt = 0; nt < 4; ++nt) {
        const float p = __expf(s[nt][r] - newm);
        s[nt][r] = p;
        rs += p;
      }
      rs += __shfl_xor(rs, 1, 32);
      rs += __shfl_xor(rs, 2, 32);
      rs += __shfl_xor(rs, 4, 32);
      rs += __shfl_xor(rs, 8, 32);
      lrun[r] = lrun[r] * corr + rs;
#pragma unroll
      for (int nt = 0; nt < 4; ++nt) oacc[nt][r] *= corr;
    }

    // C-layout P -> A-layout via per-wave LDS round trip
#pragma unroll
    for (int nt = 0; nt < 4; ++nt)
#pragma unroll
      for (int r = 0; r < 8; ++r)
        pW[(r + halfSel * 8) * 80 + nt * 16 + c16] = (_Float16)s[nt][r];
    __syncthreads();

    // O += P * V
#pragma unroll
    for (int kc = 0; kc < 2; ++kc) {
      AF ap;
      ap.h[0] = *reinterpret_cast<const v8h*>(pRow + kc * 32 + halfSel * 8);
      ap.h[1] = *reinterpret_cast<const v8h*>(pRow + kc * 32 + 16 + halfSel * 8);
#pragma unroll
      for (int nt = 0; nt < 4; ++nt) {
        AF b;
        const _Float16* bp = ldsV + (nt * 16 + c16) * 80 + kc * 32 + halfSel * 8;
        b.h[0] = *reinterpret_cast<const v8h*>(bp);
        b.h[1] = *reinterpret_cast<const v8h*>(bp + 16);
        oacc[nt] = wmma16x16x32(ap.v, b.v, oacc[nt]);
      }
    }
  }

  // epilogue: O[(b*SEQ+n)*INNER + h*64 + dh]
  const int b = bh >> 4, h = bh & 15;
#pragma unroll
  for (int nt = 0; nt < 4; ++nt) {
#pragma unroll
    for (int r = 0; r < 8; ++r) {
      const int n = qBase + r + halfSel * 8;
      const float val = oacc[nt][r] / lrun[r];
      O[(size_t)(b * SEQ + n) * INNERX + h * 64 + nt * 16 + c16] = (_Float16)val;
    }
  }
}

// ---- kernel 5: output projection [4096,1024]x[1024,1024] -> f32 + bias --
__global__ __launch_bounds__(256)
void proj_gemm_kernel(const _Float16* __restrict__ A,
                      const _Float16* __restrict__ Wh,
                      const float* __restrict__ bias,
                      float* __restrict__ out) {
  __shared__ _Float16 ldsB[64 * 40];
  const int t = threadIdx.x;
  const int lane = t & 31, wave = t >> 5;
  const int halfSel = (lane & 16) ? 1 : 0;
  const int c16 = lane & 15;
  const int colBase = blockIdx.x * 64;
  const int rowBase = blockIdx.y * 128;

  v8f acc[4];
  for (int nt = 0; nt < 4; ++nt)
    for (int i = 0; i < 8; ++i) acc[nt][i] = 0.0f;

  const _Float16* aRow = A + (size_t)(rowBase + wave * 16 + c16) * INNERX;

  for (int kk = 0; kk < INNERX; kk += 32) {
    __syncthreads();
    {
      const int kr = t & 31;
      const int n8 = (t >> 5) * 8;
      const v8h gv = *reinterpret_cast<const v8h*>(
          Wh + (size_t)(kk + kr) * DIMX + colBase + n8);
#pragma unroll
      for (int i = 0; i < 8; ++i) ldsB[(n8 + i) * 40 + kr] = gv[i];
    }
    __syncthreads();
    if (kk + 32 < INNERX)
      __builtin_prefetch(Wh + (size_t)(kk + 32 + (t & 31)) * DIMX + colBase +
                             (t >> 5) * 8, 0, 1);

    AF a;
    a.h[0] = *reinterpret_cast<const v8h*>(aRow + kk + halfSel * 8);
    a.h[1] = *reinterpret_cast<const v8h*>(aRow + kk + 16 + halfSel * 8);
#pragma unroll
    for (int nt = 0; nt < 4; ++nt) {
      AF b;
      const _Float16* bp = ldsB + (nt * 16 + c16) * 40 + halfSel * 8;
      b.h[0] = *reinterpret_cast<const v8h*>(bp);
      b.h[1] = *reinterpret_cast<const v8h*>(bp + 16);
      acc[nt] = wmma16x16x32(a.v, b.v, acc[nt]);
    }
  }

#pragma unroll
  for (int nt = 0; nt < 4; ++nt) {
    const int col = colBase + nt * 16 + c16;
    const float bv = bias[col];
#pragma unroll
    for (int r = 0; r < 8; ++r) {
      const int row = rowBase + wave * 16 + r + halfSel * 8;
      out[(size_t)row * DIMX + col] = acc[nt][r] + bv;
    }
  }
}

// ---- launcher ------------------------------------------------------------
extern "C" void kernel_launch(void* const* d_in, const int* in_sizes, int n_in,
                              void* d_out, int out_size, void* d_ws,
                              size_t ws_size, hipStream_t stream) {
  (void)in_sizes; (void)n_in; (void)out_size; (void)ws_size;
  const float* x      = (const float*)d_in[0];
  const float* ln_g   = (const float*)d_in[1];
  const float* ln_b   = (const float*)d_in[2];
  const float* w_qkv  = (const float*)d_in[3];
  const float* b_qkv  = (const float*)d_in[4];
  const float* w_proj = (const float*)d_in[5];
  const float* b_proj = (const float*)d_in[6];
  float* out = (float*)d_out;

  char* ws = (char*)d_ws;
  size_t off = 0;
  _Float16* xn_h    = (_Float16*)(ws + off); off += (size_t)ROWS * DIMX * 2;       // 8 MiB
  _Float16* wqkv_h  = (_Float16*)(ws + off); off += (size_t)DIMX * QKV_N * 2;      // 6 MiB
  _Float16* wproj_h = (_Float16*)(ws + off); off += (size_t)INNERX * DIMX * 2;     // 2 MiB
  _Float16* q_h     = (_Float16*)(ws + off); off += (size_t)ROWS * INNERX * 2;     // 8 MiB
  _Float16* k_h     = (_Float16*)(ws + off); off += (size_t)ROWS * INNERX * 2;     // 8 MiB
  _Float16* v_h     = (_Float16*)(ws + off); off += (size_t)ROWS * INNERX * 2;     // 8 MiB
  _Float16* o_h     = (_Float16*)(ws + off); off += (size_t)ROWS * INNERX * 2;     // 8 MiB

  f32_to_f16_kernel<<<2048, 256, 0, stream>>>(w_qkv, wqkv_h, DIMX * QKV_N);
  f32_to_f16_kernel<<<2048, 256, 0, stream>>>(w_proj, wproj_h, INNERX * DIMX);
  ln_kernel<<<ROWS, 256, 0, stream>>>(x, ln_g, ln_b, xn_h);
  qkv_gemm_kernel<<<dim3(QKV_N / 64, ROWS / 128), 256, 0, stream>>>(
      xn_h, wqkv_h, b_qkv, q_h, k_h, v_h);
  attn_kernel<<<dim3(SEQ / 128, BATCH * HEADS), 256, 0, stream>>>(q_h, k_h, v_h,
                                                                  o_h);
  proj_gemm_kernel<<<dim3(DIMX / 64, ROWS / 128), 256, 0, stream>>>(
      o_h, wproj_h, b_proj, out);
}
